// Attention_org_67869073211989
// MI455X (gfx1250) — matrix operand order repaired and checked by reading, not compiled, not run on hardware
//
#include <hip/hip_runtime.h>
#include <hip/hip_bf16.h>

typedef __attribute__((ext_vector_type(16))) __bf16 v16bf;
typedef __attribute__((ext_vector_type(8)))  __bf16 v8bf;   // 16 bytes
typedef __attribute__((ext_vector_type(4)))  __bf16 v4bf;   // 8 bytes
typedef __attribute__((ext_vector_type(8)))  float  v8f;
typedef __attribute__((ext_vector_type(4)))  float  v4f;

#define BT 64   // output tile edge (M and N)
#define KT 32   // K chunk (matches wmma 16x16x32)
#define LP 40   // LDS row stride in bf16 elems: 80 B -> 16B-aligned rows, bank spread

// ---------------------------------------------------------------------------
// Fragment gathers from LDS tiles laid out [row][k], row stride LP.
// A fragment 16x32 (MxK), ISA 7.12.2:  K(e) = ((e>>3)<<4) + ((lane>>4)<<3) + (e&7)
// B fragment 32x16 (KxN):              K(e) = ((e>>3)<<4) + ((lane>>4)<<3) + (e&7)
// Both gathers are two contiguous 16B runs per lane -> ds_load_b128 pairs.
// ---------------------------------------------------------------------------
__device__ __forceinline__ v16bf frag_a(const __bf16* l, int mSub, int lane) {
  v16bf a;
  const __bf16* row = l + (mSub + (lane & 15)) * LP;
  const int kh = (lane >> 4) << 3;
#pragma unroll
  for (int e = 0; e < 16; ++e)
    a[e] = row[((e >> 3) << 4) + kh + (e & 7)];
  return a;
}

__device__ __forceinline__ v16bf frag_b(const __bf16* l, int nSub, int lane) {
  v16bf b;
  const __bf16* row = l + (nSub + (lane & 15)) * LP;
  const int kb = (lane >> 4) << 3;
#pragma unroll
  for (int e = 0; e < 16; ++e)
    b[e] = row[((e >> 3) << 4) + kb + (e & 7)];
  return b;
}

// ---------------------------------------------------------------------------
// Shared 64x64-tile mainloop. Requires: Kd % 32 == 0, lda/ldb % 8 == 0,
// A/B 16B-aligned. Staging uses the CDNA5 direct-to-LDS async path:
// one GLOBAL_LOAD_ASYNC_TO_LDS_B128 per thread per operand per chunk,
// synchronized with s_wait_asynccnt + split barrier. Out-of-range rows are
// loop-invariant, so their LDS slots are zeroed once up front.
// ---------------------------------------------------------------------------
__device__ __forceinline__ void mma_tile(
    const __bf16* __restrict__ Ab, const __bf16* __restrict__ Bb,
    int M, int N, int Kd, int lda, int ldb,
    int tid, int lane, int mSub, int nSub, int mBlock, int nBlock,
    __bf16* lA, __bf16* lB, v8f& acc0, v8f& acc1) {
  const int r   = tid >> 2;          // 0..63 tile row
  const int seg = (tid & 3) << 3;    // 0,8,16,24 k-offset
  const bool aOk = (mBlock + r) < M;
  const bool bOk = (nBlock + r) < N;
  const __bf16* aCur = Ab + (long long)(mBlock + r) * lda + seg;
  const __bf16* bCur = Bb + (long long)(nBlock + r) * ldb + seg;
  __bf16* sA = lA + r * LP + seg;
  __bf16* sB = lB + r * LP + seg;
  // Per ISA 10.2, the low 32 bits of a generic LDS address are the LDS offset.
  const unsigned ldsA = (unsigned)(unsigned long long)(void*)sA;
  const unsigned ldsB = (unsigned)(unsigned long long)(void*)sB;

  // OOB rows are never written by the async copies: zero their slots once.
  {
    const v8bf z = {};
    if (!aOk) *(v8bf*)sA = z;
    if (!bOk) *(v8bf*)sB = z;
  }

  for (int k0 = 0; k0 < Kd; k0 += KT) {
    __syncthreads();                 // prior iteration's LDS reads done
    if (aOk)
      asm volatile("global_load_async_to_lds_b128 %0, %1, off"
                   :: "v"(ldsA), "v"(aCur) : "memory");
    if (bOk)
      asm volatile("global_load_async_to_lds_b128 %0, %1, off"
                   :: "v"(ldsB), "v"(bCur) : "memory");
    asm volatile("s_wait_asynccnt 0" ::: "memory");
    __syncthreads();                 // all waves' tiles resident
    const v16bf af = frag_a(lA, mSub, lane);
    const v16bf b0 = frag_b(lB, nSub, lane);
    const v16bf b1 = frag_b(lB, nSub + 16, lane);
    acc0 = __builtin_amdgcn_wmma_f32_16x16x32_bf16(false, af, false, b0,
                                                   (short)0, acc0, false, false);
    acc1 = __builtin_amdgcn_wmma_f32_16x16x32_bf16(false, af, false, b1,
                                                   (short)0, acc1, false, false);
    aCur += KT;
    bCur += KT;
  }
}

// ---------------------------------------------------------------------------
// Generic NT GEMM: C[m,n] = outScale * sum_{it,k} A[m,k] * B[n,k]
// grid.z batches via aBatch/bBatch/cBatch; innerCount accumulated passes
// (folds the head-sum into the ctx GEMM). Writes f32 (Cf) else bf16 (Cb).
// ---------------------------------------------------------------------------
__global__ void __launch_bounds__(256)
gemm_nt_bf16(const __bf16* __restrict__ A, const __bf16* __restrict__ Bm,
             float* __restrict__ Cf, __bf16* __restrict__ Cb,
             int M, int N, int Kd, int lda, int ldb, int ldc,
             long long aBatch, long long bBatch, long long cBatch,
             int innerCount, long long aInner, long long bInner, float outScale) {
  __shared__ __bf16 lA[BT * LP];
  __shared__ __bf16 lB[BT * LP];
  const int tid  = threadIdx.x;
  const int lane = tid & 31;
  const int wave = tid >> 5;
  const int mSub = (wave >> 1) * 16;
  const int nSub = (wave & 1) * 32;
  const int mBlock = blockIdx.y * BT;
  const int nBlock = blockIdx.x * BT;
  const long long z = blockIdx.z;

  v8f acc0 = {};
  v8f acc1 = {};
  for (int it = 0; it < innerCount; ++it) {
    mma_tile(A + z * aBatch + (long long)it * aInner,
             Bm + z * bBatch + (long long)it * bInner,
             M, N, Kd, lda, ldb, tid, lane, mSub, nSub, mBlock, nBlock,
             lA, lB, acc0, acc1);
  }

  // C/D layout: VGPR r -> M = r (+8 for lanes 16-31), N = lane&15
  const int rOff = (lane >> 4) << 3;
  const int col  = lane & 15;
  float*  Cfz = Cf ? Cf + z * cBatch : nullptr;
  __bf16* Cbz = Cb ? Cb + z * cBatch : nullptr;
#pragma unroll
  for (int r = 0; r < 8; ++r) {
    const int gm = mBlock + mSub + rOff + r;
    if (gm >= M) continue;
    const int c0 = nBlock + nSub + col;
    const int c1 = c0 + 16;
    if (c0 < N) {
      const float v = acc0[r] * outScale;
      if (Cfz) Cfz[(long long)gm * ldc + c0] = v;
      else     Cbz[(long long)gm * ldc + c0] = (__bf16)v;
    }
    if (c1 < N) {
      const float v = acc1[r] * outScale;
      if (Cfz) Cfz[(long long)gm * ldc + c1] = v;
      else     Cbz[(long long)gm * ldc + c1] = (__bf16)v;
    }
  }
}

// ---------------------------------------------------------------------------
// NT GEMM with TRANSPOSED scatter epilogue (for K/Q producers).
// grid.z = head h. Row index gm = b*196 + n; output:
//   Ct[ ((b*4 + h)*outN + col)*224 + n ] = value   (bf16)
// The 196..223 pad region is pre-zeroed so downstream GEMMs run Kd=224.
// ---------------------------------------------------------------------------
__global__ void __launch_bounds__(256)
gemm_nt_bT(const __bf16* __restrict__ A, const __bf16* __restrict__ Bm,
           __bf16* __restrict__ Ct,
           int M, int N, int Kd, int lda, int ldb,
           long long bBatch, int outN) {
  __shared__ __bf16 lA[BT * LP];
  __shared__ __bf16 lB[BT * LP];
  const int tid  = threadIdx.x;
  const int lane = tid & 31;
  const int wave = tid >> 5;
  const int mSub = (wave >> 1) * 16;
  const int nSub = (wave & 1) * 32;
  const int mBlock = blockIdx.y * BT;
  const int nBlock = blockIdx.x * BT;
  const long long z = blockIdx.z;   // head

  v8f acc0 = {};
  v8f acc1 = {};
  mma_tile(A, Bm + z * bBatch, M, N, Kd, lda, ldb,
           tid, lane, mSub, nSub, mBlock, nBlock, lA, lB, acc0, acc1);

  const int rOff = (lane >> 4) << 3;
  const int col  = lane & 15;
#pragma unroll
  for (int r = 0; r < 8; ++r) {
    const int gm = mBlock + mSub + rOff + r;
    if (gm >= M) continue;
    const int b = gm / 196;
    const int n = gm - b * 196;
    const long long bh = (long long)b * 4 + z;
    const int c0 = nBlock + nSub + col;
    const int c1 = c0 + 16;
    if (c0 < N) Ct[(bh * outN + c0) * 224 + n] = (__bf16)acc0[r];
    if (c1 < N) Ct[(bh * outN + c1) * 224 + n] = (__bf16)acc1[r];
  }
}

// ---------------------------------------------------------------------------
// Elementwise helpers
// ---------------------------------------------------------------------------
__global__ void cvt_bf16(const float* __restrict__ src, __bf16* __restrict__ dst,
                         long long n4) {   // n4 = n/4, n % 4 == 0
  long long i = (long long)blockIdx.x * blockDim.x + threadIdx.x;
  const long long stride = (long long)gridDim.x * blockDim.x;
  for (; i < n4; i += stride) {
    const v4f s = *(const v4f*)(src + i * 4);
    v4bf d;
    d[0] = (__bf16)s[0]; d[1] = (__bf16)s[1];
    d[2] = (__bf16)s[2]; d[3] = (__bf16)s[3];
    *(v4bf*)(dst + i * 4) = d;
  }
}

__global__ void zero_bf16(__bf16* __restrict__ dst, long long n8) {  // n % 8 == 0
  long long i = (long long)blockIdx.x * blockDim.x + threadIdx.x;
  const long long stride = (long long)gridDim.x * blockDim.x;
  const v8bf z = {};
  for (; i < n8; i += stride) *(v8bf*)(dst + i * 8) = z;
}

// ---------------------------------------------------------------------------
// Instance-norm stats: one block per (b,h) plane of ch*960 scores.
// Softmax is shift-invariant, so only rsqrt(var+eps) is needed.
// ---------------------------------------------------------------------------
__global__ void inorm_stats(const __bf16* __restrict__ sc, float* __restrict__ inv,
                            long long plane) {
  const long long base = (long long)blockIdx.x * plane;
  float s = 0.f, s2 = 0.f;
  for (long long i = threadIdx.x; i < plane; i += blockDim.x) {
    const float v = (float)sc[base + i];
    s += v; s2 += v * v;
  }
  __shared__ float r1[256];
  __shared__ float r2[256];
  const int t = threadIdx.x;
  r1[t] = s; r2[t] = s2;
  __syncthreads();
  for (int k = 128; k > 0; k >>= 1) {
    if (t < k) { r1[t] += r1[t + k]; r2[t] += r2[t + k]; }
    __syncthreads();
  }
  if (t == 0) {
    const float mean = r1[0] / (float)plane;
    const float var  = r2[0] / (float)plane - mean * mean;
    inv[blockIdx.x] = rsqrtf(fmaxf(var, 0.f) + 1e-5f);
  }
}

// ---------------------------------------------------------------------------
// Row softmax (in place, bf16). blockIdx.x = (b*H + h)*ch + d ; row len 960.
// ---------------------------------------------------------------------------
__global__ void inorm_softmax(__bf16* __restrict__ sc, const float* __restrict__ inv,
                              int ch) {
  const int KVC = 960;
  const long long row = blockIdx.x;
  const int bh = (int)(row / ch);
  const float scale = inv[bh];
  __bf16* p = sc + row * KVC;
  const int t = threadIdx.x;
  float v[4];
#pragma unroll
  for (int i = 0; i < 4; ++i) {
    const int idx = t + i * 256;
    v[i] = (idx < KVC) ? (float)p[idx] * scale : -3.0e38f;
  }
  __shared__ float red[256];
  float m = fmaxf(fmaxf(v[0], v[1]), fmaxf(v[2], v[3]));
  red[t] = m;
  __syncthreads();
  for (int k = 128; k > 0; k >>= 1) {
    if (t < k) red[t] = fmaxf(red[t], red[t + k]);
    __syncthreads();
  }
  m = red[0];
  __syncthreads();
  float sum = 0.f;
#pragma unroll
  for (int i = 0; i < 4; ++i) {
    const int idx = t + i * 256;
    v[i] = (idx < KVC) ? __expf(v[i] - m) : 0.f;
    sum += v[i];
  }
  red[t] = sum;
  __syncthreads();
  for (int k = 128; k > 0; k >>= 1) {
    if (t < k) red[t] += red[t + k];
    __syncthreads();
  }
  const float rinv = 1.f / red[0];
#pragma unroll
  for (int i = 0; i < 4; ++i) {
    const int idx = t + i * 256;
    if (idx < KVC) p[idx] = (__bf16)(v[i] * rinv);
  }
}

// ---------------------------------------------------------------------------
// Host orchestration
// ---------------------------------------------------------------------------
extern "C" void kernel_launch(void* const* d_in, const int* in_sizes, int n_in,
                              void* d_out, int out_size, void* d_ws, size_t ws_size,
                              hipStream_t stream) {
  (void)in_sizes; (void)n_in; (void)out_size; (void)ws_size;
  // setup_inputs() dict order: emb0..3, emb_all, (Wq0,Wo0)..(Wq3,Wo3), Wk, Wv
  const float* emb[4] = {(const float*)d_in[0], (const float*)d_in[1],
                         (const float*)d_in[2], (const float*)d_in[3]};
  const float* emball = (const float*)d_in[4];
  const float* Wq[4]  = {(const float*)d_in[5], (const float*)d_in[7],
                         (const float*)d_in[9], (const float*)d_in[11]};
  const float* Wo[4]  = {(const float*)d_in[6], (const float*)d_in[8],
                         (const float*)d_in[10], (const float*)d_in[12]};
  const float* WkP = (const float*)d_in[13];
  const float* WvP = (const float*)d_in[14];
  float* out = (float*)d_out;

  const int Bb = 16, KVC = 960, Hh = 4, BN = 3136;  // BN = 16*196
  const int NP = 224;                               // 196 padded to 7*32
  const int chs[4] = {64, 128, 256, 512};
  const int CHMAX = 512;

  // Workspace carve (256B aligned); per-scale buffers reused.
  char* ws = (char*)d_ws;
  size_t cur = 0;
  auto carve = [&](size_t bytes) -> char* {
    char* p = ws + cur;
    cur += (bytes + 255) & ~(size_t)255;
    return p;
  };
  __bf16* emball_bf = (__bf16*)carve((size_t)BN * KVC * 2);
  __bf16* Wk_bf     = (__bf16*)carve((size_t)Hh * KVC * KVC * 2);
  __bf16* Wv_bf     = (__bf16*)carve((size_t)Hh * KVC * KVC * 2);
  __bf16* KTb       = (__bf16*)carve((size_t)Bb * Hh * KVC * NP * 2);  // [bh][j][224]
  __bf16* Vb        = (__bf16*)carve((size_t)Hh * BN * KVC * 2);       // [h][bn][j]
  __bf16* emb_bf    = (__bf16*)carve((size_t)BN * CHMAX * 2);
  __bf16* Wq_bf     = (__bf16*)carve((size_t)Hh * CHMAX * CHMAX * 2);
  __bf16* Wo_bf     = (__bf16*)carve((size_t)CHMAX * CHMAX * 2);
  __bf16* QTb       = (__bf16*)carve((size_t)Bb * Hh * CHMAX * NP * 2); // [bh][d][224]
  __bf16* Sc        = (__bf16*)carve((size_t)Bb * Hh * CHMAX * KVC * 2);
  __bf16* Ctx       = (__bf16*)carve((size_t)BN * CHMAX * 2);
  float*  invb      = (float*)carve(64 * sizeof(float));

  auto cvt = [&](const float* s, __bf16* d, long long n) {
    const long long n4 = n >> 2;
    int blocks = (int)((n4 + 255) / 256);
    if (blocks > 4096) blocks = 4096;
    cvt_bf16<<<dim3(blocks), dim3(256), 0, stream>>>(s, d, n4);
  };
  auto zfill = [&](__bf16* d, long long n) {
    const long long n8 = n >> 3;
    int blocks = (int)((n8 + 255) / 256);
    if (blocks > 4096) blocks = 4096;
    zero_bf16<<<dim3(blocks), dim3(256), 0, stream>>>(d, n8);
  };

  cvt(emball, emball_bf, (long long)BN * KVC);
  cvt(WkP, Wk_bf, (long long)Hh * KVC * KVC);
  cvt(WvP, Wv_bf, (long long)Hh * KVC * KVC);
  zfill(KTb, (long long)Bb * Hh * KVC * NP);   // pad region must be zero

  // KT[bh][j][n] = sum_k emb_all[bn,k] Wk[h][j,k]   (transposed scatter)
  {
    dim3 g(KVC / 64, (BN + 63) / 64, Hh);
    gemm_nt_bT<<<g, 256, 0, stream>>>(
        emball_bf, Wk_bf, KTb,
        BN, KVC, KVC, KVC, KVC,
        (long long)KVC * KVC, KVC);
  }
  // V[h][bn][j] = emb_all x Wv[h]^T  (normal layout, consumed as ctx's A)
  {
    dim3 g(KVC / 64, (BN + 63) / 64, Hh);
    gemm_nt_bf16<<<g, 256, 0, stream>>>(
        emball_bf, Wv_bf, nullptr, Vb,
        BN, KVC, KVC, KVC, KVC, KVC,
        0LL, (long long)KVC * KVC, (long long)BN * KVC,
        1, 0LL, 0LL, 1.0f);
  }

  const float sscale = 0.032274861218395140f;  // 1/sqrt(960)
  long long outOff = 0;
  for (int s4 = 0; s4 < 4; ++s4) {
    const int ch = chs[s4];
    cvt(emb[s4], emb_bf, (long long)BN * ch);
    cvt(Wq[s4], Wq_bf, (long long)Hh * ch * ch);
    cvt(Wo[s4], Wo_bf, (long long)ch * ch);
    zfill(QTb, (long long)Bb * Hh * ch * NP);

    // QT[bh][d][n] = sum_c emb[bn,c] Wq[h][d,c]   (transposed scatter)
    {
      dim3 g((ch + 63) / 64, (BN + 63) / 64, Hh);
      gemm_nt_bT<<<g, 256, 0, stream>>>(
          emb_bf, Wq_bf, QTb,
          BN, ch, ch, ch, ch,
          (long long)ch * ch, ch);
    }
    // scores[bh][d,j] = (1/sqrt(KV)) * sum_n QT[d,n] KT[j,n]   (pure NT, Kd=224)
    {
      dim3 g(KVC / 64, (ch + 63) / 64, Bb * Hh);
      gemm_nt_bf16<<<g, 256, 0, stream>>>(
          QTb, KTb, nullptr, Sc,
          ch, KVC, NP, NP, NP, KVC,
          (long long)ch * NP, (long long)KVC * NP, (long long)ch * KVC,
          1, 0LL, 0LL, sscale);
    }
    // instance-norm scale per (b,h) plane, then row softmax in place
    inorm_stats<<<dim3(Bb * Hh), 256, 0, stream>>>(Sc, invb, (long long)ch * KVC);
    inorm_softmax<<<dim3(Bb * Hh * ch), 256, 0, stream>>>(Sc, invb, ch);

    // ctx[b][n,d] = (1/H) * sum_h sum_j V[h][b,n,j] * probs[b,h][d,j]
    {
      dim3 g((ch + 63) / 64, (196 + 63) / 64, Bb);
      gemm_nt_bf16<<<g, 256, 0, stream>>>(
          Vb, Sc, nullptr, Ctx,
          196, ch, KVC, KVC, KVC, ch,
          (long long)196 * KVC, (long long)Hh * ch * KVC, (long long)196 * ch,
          Hh, (long long)BN * KVC, (long long)ch * KVC, 0.25f);
    }
    // out[bn][e] = sum_d ctx[bn,d] * Wo[e,d]  (f32 output)
    {
      dim3 g((ch + 63) / 64, (BN + 63) / 64, 1);
      gemm_nt_bf16<<<g, 256, 0, stream>>>(
          Ctx, Wo_bf, out + outOff, nullptr,
          BN, ch, ch, ch, ch, ch,
          0LL, 0LL, 0LL,
          1, 0LL, 0LL, 1.0f);
    }
    outOff += (long long)BN * ch;
  }
}
